// VisionTransformerMultiView_59141699666180
// MI455X (gfx1250) — matrix operand order repaired and checked by reading, not compiled, not run on hardware
//
#include <hip/hip_runtime.h>
#include <hip/hip_bf16.h>
#include <math.h>

#define B_     2
#define NV_    8
#define T_     196
#define D_     768
#define H_     12
#define HD_    64
#define DEPTH_ 6
#define MLP_   3072
#define K2_    16
#define NC_    1000
#define N_     1568      // NV_*T_
#define NPAD_  1600      // 25 * 64
#define KBLK_  25
#define MROWS  3136      // B_*N_
#define BH_    24        // B_*H_

typedef _Float16 v16h  __attribute__((ext_vector_type(16)));
typedef _Float16 half8 __attribute__((ext_vector_type(8)));
typedef _Float16 half4 __attribute__((ext_vector_type(4)));
typedef float    v8f   __attribute__((ext_vector_type(8)));
typedef float    fl4   __attribute__((ext_vector_type(4)));

typedef unsigned int u32x4 __attribute__((ext_vector_type(4)));
typedef int          i32x4 __attribute__((ext_vector_type(4)));
typedef int          i32x8 __attribute__((ext_vector_type(8)));

enum { EPI_BIAS = 0, EPI_BIAS_RES = 1, EPI_BIAS_GELU = 2 };

__device__ __forceinline__ v8f v8f_zero() {
  v8f z;
#pragma unroll
  for (int i = 0; i < 8; ++i) z[i] = 0.0f;
  return z;
}

// ---------------------------------------------------------------------------
// TDM: load a 64x64-half 2D tile (row length 64 elems, stride0 elems between
// rows) into LDS with row padding 64 -> 72 halfs (pad_interval=32 DWORDs,
// pad_amount=4 DWORDs).  Rows >= tdim1 are OOB -> zero-filled by the TDM.
// D# layout per CDNA5 ISA 8.3/8.4.  Issue from ONE wave only.
// ---------------------------------------------------------------------------
__device__ __forceinline__ void tdm_load_tile64(unsigned lds_addr,
                                                const void* gptr,
                                                unsigned tdim1,
                                                unsigned stride0) {
  unsigned long long ga = (unsigned long long)gptr;
  u32x4 g0;
  g0[0] = 1u;                                   // count=1 (valid), user mode
  g0[1] = lds_addr;                             // LDS byte address
  g0[2] = (unsigned)(ga & 0xFFFFFFFFu);         // global_addr[31:0]
  g0[3] = (unsigned)((ga >> 32) & 0x01FFFFFFu)  // global_addr[56:32]
          | (2u << 30);                         // type = 2 ("image")
  i32x8 g1;
  g1[0] = (int)((1u << 16)        // data_size = 1 -> 2 bytes
                | (1u << 20)      // pad_enable
                | (4u << 22)      // pad_interval: code 4 = 32 DWORDs (64 halfs)
                | (3u << 25));    // pad_amount:  code 3 = 4 DWORDs (8 halfs)
  g1[1] = (int)(64u << 16);                               // tensor_dim0 = 64
  g1[2] = (int)((tdim1 & 0xFFFFu) << 16);                 // tensor_dim1 lo
  g1[3] = (int)(((tdim1 >> 16) & 0xFFFFu) | (64u << 16)); // tile_dim0 = 64
  g1[4] = (int)64u;                                       // tile_dim1 = 64
  g1[5] = (int)stride0;                                   // dim0 stride lo32
  g1[6] = 0;
  g1[7] = 0;
  i32x4 g2 = {0, 0, 0, 0};
  i32x4 g3 = {0, 0, 0, 0};
#if defined(__clang_major__) && (__clang_major__ >= 23)
  i32x8 g4 = {0, 0, 0, 0, 0, 0, 0, 0};
  __builtin_amdgcn_tensor_load_to_lds(g0, g1, g2, g3, g4, 0);
#else
  __builtin_amdgcn_tensor_load_to_lds(g0, g1, g2, g3, 0);
#endif
}

// A-fragment (16x32 f16, MxK): lane r=lane&15 -> row; s=lane>>4;
// v[0..7] = K 8s..8s+7, v[8..15] = K 16+8s..16+8s+7  (per ISA 16-bit A table)
__device__ __forceinline__ v16h frag_ld_a(const _Float16* tile, int ldt) {
  const int lane = threadIdx.x & 31;
  const int r = lane & 15, s = lane >> 4;
  const _Float16* p = tile + r * ldt + 8 * s;
  half8 lo = *(const half8*)p;
  half8 hi = *(const half8*)(p + 16);
  v16h out;
#pragma unroll
  for (int i = 0; i < 8; ++i) { out[i] = lo[i]; out[i + 8] = hi[i]; }
  return out;
}

// B-fragment (32x16 f16, KxN) stored in LDS as tile[n][k]:
// lane r -> column n; v[0..7] = K 8s..8s+7, v[8..15] = K 16+8s..16+8s+7
__device__ __forceinline__ v16h frag_ld_b(const _Float16* tile, int ldt) {
  const int lane = threadIdx.x & 31;
  const int r = lane & 15, s = lane >> 4;
  const _Float16* p = tile + r * ldt + 8 * s;
  half8 lo = *(const half8*)p;
  half8 hi = *(const half8*)(p + 16);
  v16h out;
#pragma unroll
  for (int i = 0; i < 8; ++i) { out[i] = lo[i]; out[i + 8] = hi[i]; }
  return out;
}

__device__ __forceinline__ v8f wmma_f16(v16h a, v16h b, v8f c) {
  return __builtin_amdgcn_wmma_f32_16x16x32_f16(false, a, false, b, (short)0, c,
                                                false, false);
}

// ---------------------------------------------------------------------------
// Generic WMMA GEMM: C[MxN] = A[MxK] @ B[KxN] + bias (+gelu | +residual)
// 128x128 tile per 256-thread workgroup, 8 waves in 4(M)x2(N), 32x64 per wave.
// ---------------------------------------------------------------------------
#define LDA_S 40  // 32 + 8 halfs padding
#define LDB_S 40

template <int EPI>
__global__ __launch_bounds__(256) void gemm_wmma_kernel(
    const float* __restrict__ A, const float* __restrict__ Bw,
    const float* __restrict__ bias, const float* __restrict__ res,
    float* __restrict__ C, int M, int K, int Nc) {
  __shared__ __attribute__((aligned(16))) _Float16 As[128 * LDA_S];
  __shared__ __attribute__((aligned(16))) _Float16 Bs[128 * LDB_S];
  const int tid = threadIdx.x;
  const int wid = tid >> 5;
  const int wm = wid >> 1;  // 0..3
  const int wn = wid & 1;   // 0..1
  const int mb = blockIdx.y, nb = blockIdx.x;
  const int row0 = mb * 128, col0 = nb * 128;

  v8f acc[2][4];
#pragma unroll
  for (int i = 0; i < 2; ++i)
#pragma unroll
    for (int j = 0; j < 4; ++j) acc[i][j] = v8f_zero();

  for (int kb = 0; kb < K; kb += 32) {
    // stage A tile: 128 rows x 32 cols, f32 -> f16
#pragma unroll
    for (int it = 0; it < 4; ++it) {
      int idx = tid + 256 * it;          // 0..1023 float4s
      int ar = idx >> 3;                 // 0..127
      int ac4 = (idx & 7) * 4;           // 0..28
      int grow = row0 + ar;
      half4 hv;
      if (grow < M) {
        fl4 fv = *(const fl4*)(A + (size_t)grow * K + kb + ac4);
#pragma unroll
        for (int u = 0; u < 4; ++u) hv[u] = (_Float16)fv[u];
      } else {
#pragma unroll
        for (int u = 0; u < 4; ++u) hv[u] = (_Float16)0.0f;
      }
      *(half4*)(As + ar * LDA_S + ac4) = hv;
    }
    // stage B tile transposed: Bs[n][k] = Bw[kb+k][col0+n]
#pragma unroll
    for (int it = 0; it < 4; ++it) {
      int idx = tid + 256 * it;
      int k = idx >> 5;                  // 0..31
      int nc4 = (idx & 31) * 4;          // 0..124
      fl4 fv = *(const fl4*)(Bw + (size_t)(kb + k) * Nc + col0 + nc4);
#pragma unroll
      for (int u = 0; u < 4; ++u) Bs[(nc4 + u) * LDB_S + k] = (_Float16)fv[u];
    }
    __syncthreads();

    v16h afr[2], bfr[4];
    afr[0] = frag_ld_a(As + (wm * 32 + 0) * LDA_S, LDA_S);
    afr[1] = frag_ld_a(As + (wm * 32 + 16) * LDA_S, LDA_S);
#pragma unroll
    for (int j = 0; j < 4; ++j)
      bfr[j] = frag_ld_b(Bs + (wn * 64 + 16 * j) * LDB_S, LDB_S);
#pragma unroll
    for (int i = 0; i < 2; ++i)
#pragma unroll
      for (int j = 0; j < 4; ++j) acc[i][j] = wmma_f16(afr[i], bfr[j], acc[i][j]);
    __syncthreads();
  }

  // epilogue: C layout: lane r -> col, rows = 8*s + e
  const int lane = tid & 31, r = lane & 15, s = lane >> 4;
#pragma unroll
  for (int i = 0; i < 2; ++i) {
#pragma unroll
    for (int j = 0; j < 4; ++j) {
      int col = col0 + wn * 64 + j * 16 + r;
      float bb = bias ? bias[col] : 0.0f;
#pragma unroll
      for (int e = 0; e < 8; ++e) {
        int row = row0 + wm * 32 + i * 16 + s * 8 + e;
        if (row < M) {
          float v = acc[i][j][e] + bb;
          if (EPI == EPI_BIAS_GELU)
            v = 0.5f * v * (1.0f + erff(v * 0.70710678118654752f));
          if (EPI == EPI_BIAS_RES) v += res[(size_t)row * Nc + col];
          C[(size_t)row * Nc + col] = v;
        }
      }
    }
  }
}

// ---------------------------------------------------------------------------
// LayerNorm over rows of 768
// ---------------------------------------------------------------------------
__global__ __launch_bounds__(256) void layernorm_kernel(
    const float* __restrict__ X, const float* __restrict__ g,
    const float* __restrict__ bta, float* __restrict__ Y) {
  __shared__ float red[256];
  const int row = blockIdx.x, tid = threadIdx.x;
  const float* xr = X + (size_t)row * D_;
  float s = 0.0f;
  for (int d = tid; d < D_; d += 256) s += xr[d];
  red[tid] = s;
  __syncthreads();
  for (int st = 128; st > 0; st >>= 1) {
    if (tid < st) red[tid] += red[tid + st];
    __syncthreads();
  }
  float mu = red[0] * (1.0f / D_);
  __syncthreads();
  float v = 0.0f;
  for (int d = tid; d < D_; d += 256) {
    float t = xr[d] - mu;
    v += t * t;
  }
  red[tid] = v;
  __syncthreads();
  for (int st = 128; st > 0; st >>= 1) {
    if (tid < st) red[tid] += red[tid + st];
    __syncthreads();
  }
  float rstd = rsqrtf(red[0] * (1.0f / D_) + 1e-6f);
  for (int d = tid; d < D_; d += 256)
    Y[(size_t)row * D_ + d] = (xr[d] - mu) * rstd * g[d] + bta[d];
}

// ---------------------------------------------------------------------------
// iRq = inv(SE3) = [R^T, -R^T t; 0 1]
// ---------------------------------------------------------------------------
__global__ void compute_irq_kernel(const float* __restrict__ se3q,
                                   float* __restrict__ irq) {
  int m = blockIdx.x * blockDim.x + threadIdx.x;
  if (m >= B_ * NV_) return;
  const float* A = se3q + m * 16;
  float* Oi = irq + m * 16;
#pragma unroll
  for (int i = 0; i < 3; ++i) {
#pragma unroll
    for (int j = 0; j < 3; ++j) Oi[i * 4 + j] = A[j * 4 + i];
    Oi[i * 4 + 3] =
        -(A[0 * 4 + i] * A[3] + A[1 * 4 + i] * A[7] + A[2 * 4 + i] * A[11]);
  }
  Oi[12] = 0.0f; Oi[13] = 0.0f; Oi[14] = 0.0f; Oi[15] = 1.0f;
}

// ---------------------------------------------------------------------------
// Forward GTA transforms: qkv (f32, [B,N,3,H,64]) ->
//   QT, KT (f16, [BH,N,64] row-major)  and
//   VTt   (f16, [BH,KBLK,64c,64j] block-transposed, TDM-friendly; key tail
//          zero-filled so WMMA never sees poison).
// q: SE3 part uses iRq^T, SO2 part uses Pq;   k,v: Rk and Pk.
// Softmax scale 1/8 folded into QT.
// ---------------------------------------------------------------------------
__global__ __launch_bounds__(256) void gta_fwd_kernel(
    const float* __restrict__ qkv, const float* __restrict__ se3k,
    const float* __restrict__ irq, const float* __restrict__ so2q,
    const float* __restrict__ so2k, _Float16* __restrict__ QT,
    _Float16* __restrict__ KT, _Float16* __restrict__ VTt) {
  int gid = blockIdx.x * blockDim.x + threadIdx.x;
  if (gid >= BH_ * NPAD_) return;
  const int bh = gid / NPAD_, n = gid % NPAD_;
  const int b = bh / H_, h = bh % H_;
  const int kb = n >> 6, jj = n & 63;
  _Float16* vdst = VTt + (((size_t)bh * KBLK_ + kb) << 12) + jj;
  if (n >= N_) {  // zero-fill V tail columns (keys 1568..1599)
#pragma unroll
    for (int c = 0; c < 64; ++c) vdst[(size_t)c * 64] = (_Float16)0.0f;
    return;
  }
  const int nv = n / T_;
  const float* Mq = irq + (b * NV_ + nv) * 16;   // iRq
  const float* Mk = se3k + (b * NV_ + nv) * 16;  // Rk
  const float* Pqm = so2q + (size_t)(b * N_ + n) * (K2_ * 4);
  const float* Pkm = so2k + (size_t)(b * N_ + n) * (K2_ * 4);
  const size_t rowbase = (size_t)(b * N_ + n) * (3 * D_);
  const size_t obase = ((size_t)bh * N_ + n) * 64;

  for (int st = 0; st < 3; ++st) {
    const float* src = qkv + rowbase + (size_t)(st * H_ + h) * HD_;
    float x[64], y[64];
#pragma unroll
    for (int d = 0; d < 64; ++d) x[d] = src[d];
    const float* Mm = (st == 0) ? Mq : Mk;
    const bool trans = (st == 0);  // A = iRq^T for queries
#pragma unroll
    for (int c = 0; c < 8; ++c) {
#pragma unroll
      for (int i = 0; i < 4; ++i) {
        float a = 0.0f;
#pragma unroll
        for (int j = 0; j < 4; ++j)
          a += (trans ? Mm[j * 4 + i] : Mm[i * 4 + j]) * x[c * 4 + j];
        y[c * 4 + i] = a;
      }
    }
    const float* Pm = (st == 0) ? Pqm : Pkm;
#pragma unroll
    for (int kk = 0; kk < 16; ++kk) {
      float x0 = x[32 + 2 * kk], x1 = x[32 + 2 * kk + 1];
      y[32 + 2 * kk] = Pm[kk * 4 + 0] * x0 + Pm[kk * 4 + 1] * x1;
      y[32 + 2 * kk + 1] = Pm[kk * 4 + 2] * x0 + Pm[kk * 4 + 3] * x1;
    }
    if (st == 0) {
      _Float16* dst = QT + obase;
#pragma unroll
      for (int d = 0; d < 64; ++d) dst[d] = (_Float16)(y[d] * 0.125f);
    } else if (st == 1) {
      _Float16* dst = KT + obase;
#pragma unroll
      for (int d = 0; d < 64; ++d) dst[d] = (_Float16)y[d];
    } else {  // V: block-transposed layout [c][j]
#pragma unroll
      for (int c = 0; c < 64; ++c) vdst[(size_t)c * 64] = (_Float16)y[c];
    }
  }
}

// ---------------------------------------------------------------------------
// Inverse GTA transforms on attention output: O (f32 [BH,N,64]) -> A [B,N,768]
// SE3 part: iRq;  SO2 part: Pq^T.
// ---------------------------------------------------------------------------
__global__ __launch_bounds__(256) void gta_inv_kernel(
    const float* __restrict__ O, const float* __restrict__ irq,
    const float* __restrict__ so2q, float* __restrict__ Aout) {
  int gid = blockIdx.x * blockDim.x + threadIdx.x;
  if (gid >= BH_ * N_) return;
  const int bh = gid / N_, n = gid % N_;
  const int b = bh / H_, h = bh % H_;
  const int nv = n / T_;
  const float* Mi = irq + (b * NV_ + nv) * 16;
  const float* Pm = so2q + (size_t)(b * N_ + n) * (K2_ * 4);
  const float* src = O + (size_t)gid * 64;
  float x[64], y[64];
#pragma unroll
  for (int d = 0; d < 64; ++d) x[d] = src[d];
#pragma unroll
  for (int c = 0; c < 8; ++c) {
#pragma unroll
    for (int i = 0; i < 4; ++i) {
      float a = 0.0f;
#pragma unroll
      for (int j = 0; j < 4; ++j) a += Mi[i * 4 + j] * x[c * 4 + j];
      y[c * 4 + i] = a;
    }
  }
#pragma unroll
  for (int kk = 0; kk < 16; ++kk) {
    float x0 = x[32 + 2 * kk], x1 = x[32 + 2 * kk + 1];
    y[32 + 2 * kk] = Pm[kk * 4 + 0] * x0 + Pm[kk * 4 + 2] * x1;  // P^T
    y[32 + 2 * kk + 1] = Pm[kk * 4 + 1] * x0 + Pm[kk * 4 + 3] * x1;
  }
  float* dst = Aout + (size_t)(b * N_ + n) * D_ + h * HD_;
#pragma unroll
  for (int d = 0; d < 64; ++d) dst[d] = y[d];
}

// ---------------------------------------------------------------------------
// Flash-style attention: per (query-block of 64, bh). 4 waves, wave = 16 rows.
// Q/K/V tiles staged into padded LDS by the Tensor Data Mover (wave 0 issues
// descriptors, waits on TENSORcnt); online softmax in LDS; S and PV products
// via WMMA; ragged key tail masked by TDM OOB zero-fill + explicit p=0.
// ---------------------------------------------------------------------------
__global__ __launch_bounds__(128) void attention_kernel(
    const _Float16* __restrict__ QT, const _Float16* __restrict__ KT,
    const _Float16* __restrict__ VTt, float* __restrict__ O) {
  __shared__ __attribute__((aligned(16))) _Float16 Qs[64 * 72];
  __shared__ __attribute__((aligned(16))) _Float16 Ks[64 * 72];  // [n][c]
  __shared__ __attribute__((aligned(16))) _Float16 Vs[64 * 72];  // [c][j]
  __shared__ __attribute__((aligned(16))) _Float16 Ps[64 * 72];  // [m][j]
  __shared__ float Ss[64 * 68];
  __shared__ float mstat[64], lstat[64], astat[64];

  const int tid = threadIdx.x;
  const int wid = tid >> 5, lane = tid & 31;
  const int r = lane & 15, s = lane >> 4;
  const int qb = blockIdx.x;   // 0..24
  const int bh = blockIdx.y;   // 0..23
  const size_t base = (size_t)bh * N_ * 64;

  if (wid == 0) {  // TDM: stage Q tile (tail query rows OOB -> zeros)
    unsigned qv = (unsigned)(N_ - qb * 64);
    if (qv > 64u) qv = 64u;
    tdm_load_tile64((unsigned)(size_t)&Qs[0],
                    QT + base + (size_t)qb * 64 * 64, qv, 64u);
  }
  if (tid < 64) { mstat[tid] = -INFINITY; lstat[tid] = 0.0f; }

  v8f oacc[4];
#pragma unroll
  for (int j = 0; j < 4; ++j) oacc[j] = v8f_zero();

  for (int kb = 0; kb < KBLK_; ++kb) {
    if (wid == 0) {  // TDM: stage K (row-major) and pre-transposed V
      unsigned kvv = (unsigned)(N_ - kb * 64);
      if (kvv > 64u) kvv = 64u;
      tdm_load_tile64((unsigned)(size_t)&Ks[0],
                      KT + base + (size_t)kb * 64 * 64, kvv, 64u);
      tdm_load_tile64((unsigned)(size_t)&Vs[0],
                      VTt + (((size_t)bh * KBLK_ + kb) << 12), 64u, 64u);
      __builtin_amdgcn_s_wait_tensorcnt(0);
    }
    __syncthreads();

    // S = Q @ K^T   (contraction over channel c, two k-steps of 32)
    v16h aq0 = frag_ld_a(Qs + (wid * 16) * 72 + 0, 72);
    v16h aq1 = frag_ld_a(Qs + (wid * 16) * 72 + 32, 72);
#pragma unroll
    for (int j = 0; j < 4; ++j) {
      v8f sacc = v8f_zero();
      v16h b0 = frag_ld_b(Ks + (16 * j) * 72 + 0, 72);
      v16h b1 = frag_ld_b(Ks + (16 * j) * 72 + 32, 72);
      sacc = wmma_f16(aq0, b0, sacc);
      sacc = wmma_f16(aq1, b1, sacc);
#pragma unroll
      for (int e = 0; e < 8; ++e)
        Ss[(wid * 16 + s * 8 + e) * 68 + j * 16 + r] = sacc[e];
    }
    __syncthreads();

    // online softmax: lanes 0..15 of each wave, one row each
    if (lane < 16) {
      const int row = wid * 16 + lane;
      int nvalid = N_ - kb * 64;
      if (nvalid > 64) nvalid = 64;
      float mold = mstat[row], mnew = mold;
      for (int j = 0; j < nvalid; ++j) mnew = fmaxf(mnew, Ss[row * 68 + j]);
      float alpha = __expf(mold - mnew);
      float lsum = 0.0f;
      for (int j = 0; j < 64; ++j) {
        float p = (j < nvalid) ? __expf(Ss[row * 68 + j] - mnew) : 0.0f;
        Ps[row * 72 + j] = (_Float16)p;
        lsum += p;
      }
      mstat[row] = mnew;
      lstat[row] = alpha * lstat[row] + lsum;
      astat[row] = alpha;
    }
    __syncthreads();

    // rescale running O, then O += P @ V
    float al[8];
#pragma unroll
    for (int e = 0; e < 8; ++e) al[e] = astat[wid * 16 + s * 8 + e];
#pragma unroll
    for (int j = 0; j < 4; ++j)
#pragma unroll
      for (int e = 0; e < 8; ++e) oacc[j][e] *= al[e];

    v16h ap0 = frag_ld_a(Ps + (wid * 16) * 72 + 0, 72);
    v16h ap1 = frag_ld_a(Ps + (wid * 16) * 72 + 32, 72);
#pragma unroll
    for (int j = 0; j < 4; ++j) {
      v16h b0 = frag_ld_b(Vs + (16 * j) * 72 + 0, 72);
      v16h b1 = frag_ld_b(Vs + (16 * j) * 72 + 32, 72);
      oacc[j] = wmma_f16(ap0, b0, oacc[j]);
      oacc[j] = wmma_f16(ap1, b1, oacc[j]);
    }
    __syncthreads();  // all waves done reading Ks/Vs before next TDM
  }

  // final normalization + store
  float invl[8];
#pragma unroll
  for (int e = 0; e < 8; ++e) invl[e] = 1.0f / lstat[wid * 16 + s * 8 + e];
#pragma unroll
  for (int j = 0; j < 4; ++j)
#pragma unroll
    for (int e = 0; e < 8; ++e) {
      int q = qb * 64 + wid * 16 + s * 8 + e;
      if (q < N_) O[base + (size_t)q * 64 + j * 16 + r] = oacc[j][e] * invl[e];
    }
}

// ---------------------------------------------------------------------------
// Final LN on token 0 of each batch + classification head (2x1000)
// ---------------------------------------------------------------------------
__global__ __launch_bounds__(256) void head_kernel(
    const float* __restrict__ X, const float* __restrict__ g,
    const float* __restrict__ bta, const float* __restrict__ Wh,
    const float* __restrict__ bh_, float* __restrict__ out) {
  __shared__ float sh[D_];
  __shared__ float red[256];
  const int b = blockIdx.x, tid = threadIdx.x;
  const float* xr = X + (size_t)(b * N_) * D_;  // token 0 of batch b
  float ssum = 0.0f;
  for (int d = tid; d < D_; d += 256) ssum += xr[d];
  red[tid] = ssum;
  __syncthreads();
  for (int st = 128; st > 0; st >>= 1) {
    if (tid < st) red[tid] += red[tid + st];
    __syncthreads();
  }
  float mu = red[0] * (1.0f / D_);
  __syncthreads();
  float v = 0.0f;
  for (int d = tid; d < D_; d += 256) {
    float t = xr[d] - mu;
    v += t * t;
  }
  red[tid] = v;
  __syncthreads();
  for (int st = 128; st > 0; st >>= 1) {
    if (tid < st) red[tid] += red[tid + st];
    __syncthreads();
  }
  float rstd = rsqrtf(red[0] * (1.0f / D_) + 1e-6f);
  for (int d = tid; d < D_; d += 256) sh[d] = (xr[d] - mu) * rstd * g[d] + bta[d];
  __syncthreads();
  for (int c = tid; c < NC_; c += 256) {
    float acc = bh_[c];
    for (int d = 0; d < D_; ++d) acc += sh[d] * Wh[(size_t)d * NC_ + c];
    out[b * NC_ + c] = acc;
  }
}

// ---------------------------------------------------------------------------
extern "C" void kernel_launch(void* const* d_in, const int* in_sizes, int n_in,
                              void* d_out, int out_size, void* d_ws,
                              size_t ws_size, hipStream_t stream) {
  (void)in_sizes; (void)n_in; (void)out_size; (void)ws_size;
  const float* in_x   = (const float*)d_in[0];
  const float* se3q   = (const float*)d_in[1];
  const float* se3k   = (const float*)d_in[2];
  const float* so2q   = (const float*)d_in[3];
  const float* so2k   = (const float*)d_in[4];
  const float* w_qkv  = (const float*)d_in[5];
  const float* b_qkv  = (const float*)d_in[6];
  const float* w_proj = (const float*)d_in[7];
  const float* b_proj = (const float*)d_in[8];
  const float* ln1_g  = (const float*)d_in[9];
  const float* ln1_b  = (const float*)d_in[10];
  const float* ln2_g  = (const float*)d_in[11];
  const float* ln2_b  = (const float*)d_in[12];
  const float* w_fc1  = (const float*)d_in[13];
  const float* b_fc1  = (const float*)d_in[14];
  const float* w_fc2  = (const float*)d_in[15];
  const float* b_fc2  = (const float*)d_in[16];
  const float* lnf_g  = (const float*)d_in[17];
  const float* lnf_b  = (const float*)d_in[18];
  const float* w_head = (const float*)d_in[19];
  const float* b_head = (const float*)d_in[20];

  char* ws = (char*)d_ws;
  size_t off = 0;
  auto alloc = [&](size_t bytes) {
    char* p = ws + off;
    off = (off + bytes + 255) & ~(size_t)255;
    return p;
  };
  float*    X    = (float*)alloc((size_t)MROWS * D_ * 4);
  float*    Hbuf = (float*)alloc((size_t)MROWS * D_ * 4);
  float*    BIG  = (float*)alloc((size_t)MROWS * MLP_ * 4);  // qkv & fc1 out
  _Float16* QT   = (_Float16*)alloc((size_t)BH_ * N_ * 64 * 2);
  _Float16* KT   = (_Float16*)alloc((size_t)BH_ * N_ * 64 * 2);
  _Float16* VTt  = (_Float16*)alloc((size_t)BH_ * KBLK_ * 4096 * 2);
  float*    Obuf = (float*)alloc((size_t)BH_ * N_ * 64 * 4);
  float*    Abuf = (float*)alloc((size_t)MROWS * D_ * 4);
  float*    IRQ  = (float*)alloc((size_t)B_ * NV_ * 16 * 4);

  hipMemcpyAsync(X, in_x, (size_t)MROWS * D_ * 4, hipMemcpyDeviceToDevice,
                 stream);
  compute_irq_kernel<<<1, 32, 0, stream>>>(se3q, IRQ);

  const int MT = (MROWS + 127) / 128;  // 25
  for (int i = 0; i < DEPTH_; ++i) {
    layernorm_kernel<<<MROWS, 256, 0, stream>>>(X, ln1_g + i * D_,
                                                ln1_b + i * D_, Hbuf);
    gemm_wmma_kernel<EPI_BIAS><<<dim3((3 * D_) / 128, MT), 256, 0, stream>>>(
        Hbuf, w_qkv + (size_t)i * D_ * 3 * D_, b_qkv + i * 3 * D_, nullptr, BIG,
        MROWS, D_, 3 * D_);
    gta_fwd_kernel<<<(BH_ * NPAD_ + 255) / 256, 256, 0, stream>>>(
        BIG, se3k, IRQ, so2q, so2k, QT, KT, VTt);
    attention_kernel<<<dim3(KBLK_, BH_), 128, 0, stream>>>(QT, KT, VTt, Obuf);
    gta_inv_kernel<<<(BH_ * N_ + 255) / 256, 256, 0, stream>>>(Obuf, IRQ, so2q,
                                                               Abuf);
    gemm_wmma_kernel<EPI_BIAS_RES><<<dim3(D_ / 128, MT), 256, 0, stream>>>(
        Abuf, w_proj + (size_t)i * D_ * D_, b_proj + i * D_, X, X, MROWS, D_,
        D_);
    layernorm_kernel<<<MROWS, 256, 0, stream>>>(X, ln2_g + i * D_,
                                                ln2_b + i * D_, Hbuf);
    gemm_wmma_kernel<EPI_BIAS_GELU><<<dim3(MLP_ / 128, MT), 256, 0, stream>>>(
        Hbuf, w_fc1 + (size_t)i * D_ * MLP_, b_fc1 + i * MLP_, nullptr, BIG,
        MROWS, D_, MLP_);
    gemm_wmma_kernel<EPI_BIAS_RES><<<dim3(D_ / 128, MT), 256, 0, stream>>>(
        BIG, w_fc2 + (size_t)i * MLP_ * D_, b_fc2 + i * D_, X, X, MROWS, MLP_,
        D_);
  }
  head_kernel<<<B_, 256, 0, stream>>>(X, lnf_g, lnf_b, w_head, b_head,
                                      (float*)d_out);
}
